// LocalAttentionLayer_73839077753207
// MI455X (gfx1250) — matrix operand order repaired and checked by reading, not compiled, not run on hardware
//
#include <hip/hip_runtime.h>
#include <hip/hip_bf16.h>
#include <math.h>
#include <stdint.h>

typedef __attribute__((ext_vector_type(2)))  float   v2f;
typedef __attribute__((ext_vector_type(4)))  float   f4;
typedef __attribute__((ext_vector_type(8)))  float   v8f;
typedef __attribute__((ext_vector_type(8)))  __bf16  v8bf;
typedef __attribute__((ext_vector_type(16))) __bf16  v16bf;

#define B_  8
#define L_  8192
#define D_  512
#define H_  8
#define KD  64
#define W_  128
#define C_  (L_ / W_)
#define M_  (B_ * L_)

// ---------------------------------------------------------------------------
// WMMA helpers
// ---------------------------------------------------------------------------
__device__ __forceinline__ v8f wmma4(v2f a, v2f b, v8f c) {   // exact fp32
  return __builtin_amdgcn_wmma_f32_16x16x4_f32(
      false, a, false, b, (short)0, c, false, false);
}
__device__ __forceinline__ v8f wmma_bf(v16bf a, v16bf b, v8f c) {
  return __builtin_amdgcn_wmma_f32_16x16x32_bf16(
      false, a, false, b, (short)0, c, false, false);
}
__device__ __forceinline__ void split_bf16(float x, __bf16& h, __bf16& l) {
  h = (__bf16)x;
  l = (__bf16)(x - (float)h);
}
__device__ __forceinline__ v16bf cat8(v8bf lo, v8bf hi) {
  v16bf r;
#pragma unroll
  for (int i = 0; i < 8; ++i) { r[i] = lo[i]; r[8 + i] = hi[i]; }
  return r;
}

// ---------------------------------------------------------------------------
// GEMM: C[M x 512] = alpha * A[M x 512] * B[512 x 512]  (fp32 in/out)
// Math: split-bf16 (hi/lo) on V_WMMA_F32_16X16X32_BF16, 3 WMMAs per product:
//   a*b ~= aH*bH + aH*bL + aL*bH   (lo*lo term ~2^-32, below fp32 eps)
// Block tile 128x64, BK=32, 256 threads = 8 waves, wave tile 32x32.
// ---------------------------------------------------------------------------
#define GM  128
#define GN  64
#define GK  32
#define AST 40   // bf16 row stride for A tiles  (80 B: 16B-aligned, conflict-free)
#define BST 40   // bf16 row stride for Bt tiles

__global__ __launch_bounds__(256)
void gemm_f32_bf16x3(const float* __restrict__ A, const float* __restrict__ Bm,
                     float* __restrict__ C, float alpha) {
  __shared__ __bf16 Ah[GM][AST];   // A tile, hi part  [m][kk]
  __shared__ __bf16 Al[GM][AST];   // A tile, lo part
  __shared__ __bf16 Bth[GN][BST];  // B tile transposed, hi  [n][kk]
  __shared__ __bf16 Btl[GN][BST];  // B tile transposed, lo

  const int tid  = threadIdx.x;
  const int lane = tid & 31;
  const int wave = tid >> 5;
  const int wm   = wave & 3;          // 4 waves along M
  const int wn   = wave >> 2;         // 2 waves along N
  const int lm   = lane & 15;
  const int kbA  = (lane >> 4) * 8;   // A frag K base {0,8}
  const int kbB  = (lane >> 4) * 16;  // B frag K base {0,16}
  const int mhi  = (lane >> 4) << 3;  // C/D row offset {0,8}

  const int m_blk = blockIdx.y * GM;
  const int n_blk = blockIdx.x * GN;

  v8f acc[2][2];
#pragma unroll
  for (int i = 0; i < 2; ++i)
#pragma unroll
    for (int j = 0; j < 2; ++j)
      acc[i][j] = (v8f){0.f, 0.f, 0.f, 0.f, 0.f, 0.f, 0.f, 0.f};

  for (int k0 = 0; k0 < D_; k0 += GK) {
    // --- stage A tile 128x32: load f32, split to bf16 hi/lo ---
    {
      const int c4 = (tid & 7) * 4;
      const int r  = tid >> 3;
      const float* src = A + (size_t)m_blk * D_ + k0;
#pragma unroll
      for (int p = 0; p < 4; ++p) {
        const int row = r + p * 32;
        f4 t = *(const f4*)(src + (size_t)row * D_ + c4);
#pragma unroll
        for (int e = 0; e < 4; ++e) {
          __bf16 h, l;
          split_bf16(t[e], h, l);
          Ah[row][c4 + e] = h;
          Al[row][c4 + e] = l;
        }
        if (k0 + GK < D_)
          __builtin_prefetch(src + (size_t)row * D_ + c4 + GK, 0, 0);
      }
    }
    // --- stage B tile 32x64 transposed to [n][kk] ---
    {
      const int c4 = (tid & 15) * 4;   // n
      const int r  = tid >> 4;         // kk
      const float* src = Bm + (size_t)k0 * D_ + n_blk;
#pragma unroll
      for (int p = 0; p < 2; ++p) {
        const int kk = r + p * 16;
        f4 t = *(const f4*)(src + (size_t)kk * D_ + c4);
#pragma unroll
        for (int e = 0; e < 4; ++e) {
          __bf16 h, l;
          split_bf16(t[e], h, l);
          Bth[c4 + e][kk] = h;
          Btl[c4 + e][kk] = l;
        }
        if (k0 + GK < D_)
          __builtin_prefetch(src + (size_t)(kk + GK) * D_ + c4, 0, 0);
      }
    }
    __syncthreads();

    // --- fragments (16-bit layouts per ISA 7.12.2) ---
    v16bf aH[2], aL[2], bH[2], bL[2];
#pragma unroll
    for (int i = 0; i < 2; ++i) {
      const int m = wm * 32 + i * 16 + lm;
      aH[i] = cat8(*(const v8bf*)&Ah[m][kbA], *(const v8bf*)&Ah[m][16 + kbA]);
      aL[i] = cat8(*(const v8bf*)&Al[m][kbA], *(const v8bf*)&Al[m][16 + kbA]);
    }
#pragma unroll
    for (int j = 0; j < 2; ++j) {
      const int n = wn * 32 + j * 16 + lm;
      bH[j] = cat8(*(const v8bf*)&Bth[n][kbA], *(const v8bf*)&Bth[n][16 + kbA]);
      bL[j] = cat8(*(const v8bf*)&Btl[n][kbA], *(const v8bf*)&Btl[n][16 + kbA]);
    }
#pragma unroll
    for (int i = 0; i < 2; ++i)
#pragma unroll
      for (int j = 0; j < 2; ++j) {
        acc[i][j] = wmma_bf(aH[i], bH[j], acc[i][j]);
        acc[i][j] = wmma_bf(aH[i], bL[j], acc[i][j]);
        acc[i][j] = wmma_bf(aL[i], bH[j], acc[i][j]);
      }
    __syncthreads();
  }

#pragma unroll
  for (int i = 0; i < 2; ++i)
#pragma unroll
    for (int j = 0; j < 2; ++j)
#pragma unroll
      for (int r = 0; r < 8; ++r) {
        const int m = m_blk + wm * 32 + i * 16 + mhi + r;
        const int n = n_blk + wn * 32 + j * 16 + lm;
        C[(size_t)m * D_ + n] = alpha * acc[i][j][r];
      }
}

// ---------------------------------------------------------------------------
// Local attention over one (b, chunk, head, half): 64 q rows x 129 keys.
// Q/V tiles staged via global_load_async_to_lds_b128 (ASYNCcnt path).
// ---------------------------------------------------------------------------
#define QS_STR 68
#define KT_STR 132
#define VS_STR 68
#define S_STR  132

__device__ __forceinline__ void async_ld_b128(float* lds_dst,
                                              const float* gbase,
                                              unsigned byte_off) {
  const unsigned lds_off = (unsigned)(uintptr_t)lds_dst;  // low 32b = LDS addr
  asm volatile("global_load_async_to_lds_b128 %0, %1, %2"
               :: "v"(lds_off), "v"(byte_off), "s"(gbase)
               : "memory");
}

__global__ __launch_bounds__(256)
void local_attn_wmma(const float* __restrict__ q, const float* __restrict__ k,
                     const float* __restrict__ v, float* __restrict__ ctx) {
  extern __shared__ float smem[];
  float* Qs = smem;                       // 64 x 68
  float* Kt = Qs + 64 * QS_STR;           // 64 x 132 (Kt[kdim][key])
  float* Vs = Kt + 64 * KT_STR;           // 128 x 68 (Vs[key][kdim])
  float* S  = Vs + 128 * VS_STR;          // 64 x 132 (col 128 = CLS)
  float* Kc = S + 64 * S_STR;             // 64
  float* Vc = Kc + 64;                    // 64

  const int tid  = threadIdx.x;
  const int lane = tid & 31;
  const int wave = tid >> 5;
  const int lm   = lane & 15;
  const int ko   = (lane >> 4) << 1;
  const int mhi  = (lane >> 4) << 3;

  const int h    = blockIdx.x;
  const int c    = blockIdx.y >> 1;
  const int half = blockIdx.y & 1;
  const int b    = blockIdx.z;
  const int l0   = c * W_;
  const int r0   = half * 64;
  const size_t base = (size_t)b * L_ * D_ + (size_t)h * KD;

  // --- stage Q (64x64) via async DMA to LDS ---
  {
    const int c4 = (tid & 15) * 4;
    const int r  = tid >> 4;
#pragma unroll
    for (int p = 0; p < 4; ++p) {
      const int row = r + p * 16;
      const unsigned go =
          (unsigned)((base + (size_t)(l0 + r0 + row) * D_ + c4) * sizeof(float));
      async_ld_b128(Qs + row * QS_STR + c4, q, go);
    }
  }
  // --- stage V (Vs[key][kdim], 128x64) via async DMA to LDS ---
  {
    const int c4 = (tid & 15) * 4;
    const int ky = tid >> 4;
#pragma unroll
    for (int p = 0; p < 8; ++p) {
      const int key = ky + p * 16;
      const unsigned go =
          (unsigned)((base + (size_t)(l0 + key) * D_ + c4) * sizeof(float));
      async_ld_b128(Vs + key * VS_STR + c4, v, go);
    }
  }
  // --- stage K^T (Kt[kdim][key], 64x128): VGPR-mediated transpose ---
  {
    const int c4 = (tid & 15) * 4;
    const int ky = tid >> 4;
#pragma unroll
    for (int p = 0; p < 8; ++p) {
      const int key = ky + p * 16;
      f4 t = *(const f4*)(k + base + (size_t)(l0 + key) * D_ + c4);
      Kt[(c4 + 0) * KT_STR + key] = t.x; Kt[(c4 + 1) * KT_STR + key] = t.y;
      Kt[(c4 + 2) * KT_STR + key] = t.z; Kt[(c4 + 3) * KT_STR + key] = t.w;
    }
  }
  if (tid < 64) {              // CLS key/value (sequence position 0)
    Kc[tid] = k[base + tid];
    Vc[tid] = v[base + tid];
  }
  asm volatile("s_wait_asynccnt 0" ::: "memory");
  __syncthreads();

  // --- phase 1: S[64x128] = Q * K^T via fp32 WMMA (uniform control flow) ---
  {
    const int wm  = wave & 3;
    const int wn2 = wave >> 2;
    v8f sa[4];
#pragma unroll
    for (int j = 0; j < 4; ++j)
      sa[j] = (v8f){0.f, 0.f, 0.f, 0.f, 0.f, 0.f, 0.f, 0.f};
#pragma unroll
    for (int kk = 0; kk < KD; kk += 4) {
      v2f a;
      const int m = wm * 16 + lm;
      a.x = Qs[m * QS_STR + kk + ko];
      a.y = Qs[m * QS_STR + kk + ko + 1];
#pragma unroll
      for (int j = 0; j < 4; ++j) {
        const int n = (wn2 * 4 + j) * 16 + lm;
        v2f bf;
        bf.x = Kt[(kk + ko) * KT_STR + n];
        bf.y = Kt[(kk + ko + 1) * KT_STR + n];
        sa[j] = wmma4(a, bf, sa[j]);
      }
    }
#pragma unroll
    for (int j = 0; j < 4; ++j)
#pragma unroll
      for (int r = 0; r < 8; ++r) {
        const int m = wm * 16 + mhi + r;
        const int n = (wn2 * 4 + j) * 16 + lm;
        S[m * S_STR + n] = sa[j][r];
      }
  }
  __syncthreads();

  // --- CLS column + per-row softmax over 129 ---
  if (tid < 64) {
    float sc = 0.f;
#pragma unroll
    for (int kk = 0; kk < KD; ++kk) sc += Qs[tid * QS_STR + kk] * Kc[kk];
    S[tid * S_STR + 128] = sc;
  }
  __syncthreads();
  if (tid < 64) {
    float mx = -3.402823466e+38f;
    for (int n = 0; n <= 128; ++n) mx = fmaxf(mx, S[tid * S_STR + n]);
    float sum = 0.f;
    for (int n = 0; n <= 128; ++n) {
      const float e = __expf(S[tid * S_STR + n] - mx);
      S[tid * S_STR + n] = e;
      sum += e;
    }
    const float inv = 1.f / sum;
    for (int n = 0; n <= 128; ++n) S[tid * S_STR + n] *= inv;
  }
  __syncthreads();

  // --- phase 2: ctx[64x64] = P * V + p_cls (x) V_cls ---
  {
    const int wm = wave & 3;
    const int wn = wave >> 2;
    v8f ca[2];
#pragma unroll
    for (int j = 0; j < 2; ++j)
      ca[j] = (v8f){0.f, 0.f, 0.f, 0.f, 0.f, 0.f, 0.f, 0.f};
#pragma unroll
    for (int kk = 0; kk < W_; kk += 4) {
      v2f a;
      const int m = wm * 16 + lm;
      a.x = S[m * S_STR + kk + ko];
      a.y = S[m * S_STR + kk + ko + 1];
#pragma unroll
      for (int j = 0; j < 2; ++j) {
        const int n = (wn * 2 + j) * 16 + lm;
        v2f bf;
        bf.x = Vs[(kk + ko) * VS_STR + n];
        bf.y = Vs[(kk + ko + 1) * VS_STR + n];
        ca[j] = wmma4(a, bf, ca[j]);
      }
    }
#pragma unroll
    for (int j = 0; j < 2; ++j)
#pragma unroll
      for (int r = 0; r < 8; ++r) {
        const int m = wm * 16 + mhi + r;
        const int n = (wn * 2 + j) * 16 + lm;
        const float val = ca[j][r] + S[m * S_STR + 128] * Vc[n];
        ctx[base + (size_t)(l0 + r0 + m) * D_ + n] = val;
      }
  }
}

// ---------------------------------------------------------------------------
// Host launcher
// ---------------------------------------------------------------------------
extern "C" void kernel_launch(void* const* d_in, const int* in_sizes, int n_in,
                              void* d_out, int out_size, void* d_ws, size_t ws_size,
                              hipStream_t stream) {
  (void)in_sizes; (void)n_in; (void)out_size; (void)ws_size;
  const float* xq  = (const float*)d_in[0];  // [B,L,D]
  const float* xkv = (const float*)d_in[1];  // [B,L,D]
  const float* Wq  = (const float*)d_in[2];  // [D,H,K] -> [512,512]
  const float* Wk  = (const float*)d_in[3];
  const float* Wv  = (const float*)d_in[4];
  const float* Wo  = (const float*)d_in[5];  // [H,K,D] -> [512,512]
  float* out = (float*)d_out;

  const size_t QN = (size_t)M_ * D_;
  float* qb = (float*)d_ws;
  float* kb = qb + QN;
  float* vb = kb + QN;
  float* cb = vb + QN;

  const dim3 gblk(256);
  const dim3 ggrid(D_ / GN, M_ / GM);        // (8, 512)
  const float scale = 0.125f;                // 1/sqrt(64)

  gemm_f32_bf16x3<<<ggrid, gblk, 0, stream>>>(xq,  Wq, qb, scale);
  gemm_f32_bf16x3<<<ggrid, gblk, 0, stream>>>(xkv, Wk, kb, 1.0f);
  gemm_f32_bf16x3<<<ggrid, gblk, 0, stream>>>(xkv, Wv, vb, 1.0f);

  const size_t smem = (size_t)(64 * QS_STR + 64 * KT_STR + 128 * VS_STR +
                               64 * S_STR + 128) * sizeof(float);  // ~120 KB
  const dim3 agrid(H_, C_ * 2, B_);          // (8, 128, 8)
  local_attn_wmma<<<agrid, gblk, smem, stream>>>(qb, kb, vb, cb);

  gemm_f32_bf16x3<<<ggrid, gblk, 0, stream>>>(cb, Wo, out, 1.0f);
}
